// GCN_60833916781039
// MI455X (gfx1250) — matrix operand order repaired and checked by reading, not compiled, not run on hardware
//
#include <hip/hip_runtime.h>
#include <hip/hip_bf16.h>

// ---------------------------------------------------------------------------
// GCN forward for MI455X (gfx1250, wave32).
//   - GEMMs: v_wmma_f32_16x16x32_bf16 with fp32->bf16 hi/lo split (3 WMMAs per
//     K=32 tile) for near-fp32 accuracy at bf16-WMMA rate.
//     B fragments batched per K-block + round-robin WMMA issue (dep distance
//     NT) to pipeline the accumulate chain; all B addresses are compile-time
//     immediate offsets off one base pointer (hi at 0, lo at +K*C).
//   - Edge aggregation: L2-resident gather + global_atomic_add_f32 scatter
//     (activations [N,256] = 102MB fit in the 192MB L2).
//   - BN+ReLU fused into the A-operand load of the following GEMM (vectorized
//     scale/shift loads).
//   - agg_b0/agg_b1 cancel inside BatchNorm (BN(x+c)==BN(x)) and are skipped.
// ---------------------------------------------------------------------------

typedef __bf16 bf16_t;
typedef __attribute__((ext_vector_type(16))) __bf16 v16bf;
typedef __attribute__((ext_vector_type(8)))  float  v8f;
typedef __attribute__((ext_vector_type(4)))  float  v4f;

#define NNODES 100000
#define NEDGES 3200000
#define DIN    128
#define HDIM   256
#define EMBD   64
#define BN_EPS 1e-5f

// --- weight prep: fp32 [K][C] row-major -> transposed bf16 hi/lo [C][K] -----
// region layout (bf16 elems): [w0h 32768][w0l 32768][w1h 65536][w1l 65536]
//                             [woh 16384][wol 16384]   (lo always at +K*C)
__global__ void prep_weights(const float* __restrict__ W0,
                             const float* __restrict__ W1,
                             const float* __restrict__ Wo,
                             bf16_t* __restrict__ out) {
  int i = blockIdx.x * blockDim.x + threadIdx.x;
  const float* src; int K, C, base, li;
  if (i < 32768)              { src = W0; K = 128; C = 256; base = 0;      li = i;          }
  else if (i < 32768 + 65536) { src = W1; K = 256; C = 256; base = 65536;  li = i - 32768;  }
  else if (i < 114688)        { src = Wo; K = 256; C = 64;  base = 196608; li = i - 98304;  }
  else return;
  int k = li / C, n = li % C;
  float v = src[li];
  bf16_t hi = (bf16_t)v;
  float  lo = v - (float)hi;
  out[base + n * K + k]         = hi;
  out[base + K * C + n * K + k] = (bf16_t)lo;
}

// --- WMMA GEMM: Cout[M, C] = op(A[M,K]) @ B[K, C] (+bias) -------------------
// op(A) = optional per-K-column BN scale/shift + ReLU (fused from prev layer).
// One wave per 16-row tile; NT 16-col tiles per wave; grid.y = column groups.
// B layout: transposed bf16, hi part at B, lo part at B + K*C.
template <int K, int C, int NT, bool BN, bool BIAS>
__global__ __launch_bounds__(32)
void gemm_wmma(const float*  __restrict__ A,
               const bf16_t* __restrict__ B,
               const float*  __restrict__ bias,
               const float*  __restrict__ bnscale,
               const float*  __restrict__ bnshift,
               float* __restrict__ Cout) {
  const int lane    = threadIdx.x & 31;
  const int m       = lane & 15;   // row within tile (A) / col within tile (B,C)
  const int half    = lane >> 4;
  const int tile    = blockIdx.x;
  const int colbase = blockIdx.y * (NT * 16);

  const float*  Abase = A + (size_t)(tile * 16 + m) * K + half * 16;
  const bf16_t* Bbase = B + (size_t)(colbase + m) * K + half * 16;

  v8f acc[NT];
  v8f zero = {0.f, 0.f, 0.f, 0.f, 0.f, 0.f, 0.f, 0.f};
#pragma unroll
  for (int t = 0; t < NT; ++t) acc[t] = zero;

#pragma unroll
  for (int kb = 0; kb < K; kb += 32) {
    // A fragment (contiguous k order, matches B): one contiguous 16-float run.
    const float* ap = Abase + kb;
    v4f a0 = *(const v4f*)(ap);
    v4f a1 = *(const v4f*)(ap + 4);
    v4f a2 = *(const v4f*)(ap + 8);
    v4f a3 = *(const v4f*)(ap + 12);
    float fa[16];
#pragma unroll
    for (int i = 0; i < 4; ++i) {
      fa[i] = a0[i]; fa[4 + i] = a1[i]; fa[8 + i] = a2[i]; fa[12 + i] = a3[i];
    }
    if (BN) {
      const float* sp = bnscale + kb + half * 16;
      const float* hp = bnshift + kb + half * 16;
      v4f s0 = *(const v4f*)(sp),      s1 = *(const v4f*)(sp + 4);
      v4f s2 = *(const v4f*)(sp + 8),  s3 = *(const v4f*)(sp + 12);
      v4f h0 = *(const v4f*)(hp),      h1 = *(const v4f*)(hp + 4);
      v4f h2 = *(const v4f*)(hp + 8),  h3 = *(const v4f*)(hp + 12);
      float fs[16], fh[16];
#pragma unroll
      for (int i = 0; i < 4; ++i) {
        fs[i] = s0[i]; fs[4 + i] = s1[i]; fs[8 + i] = s2[i]; fs[12 + i] = s3[i];
        fh[i] = h0[i]; fh[4 + i] = h1[i]; fh[8 + i] = h2[i]; fh[12 + i] = h3[i];
      }
#pragma unroll
      for (int j = 0; j < 16; ++j) {
        float v = fmaf(fa[j], fs[j], fh[j]);
        fa[j] = v > 0.f ? v : 0.f;
      }
    }
    // fp32 -> bf16 hi/lo split
    v16bf ah, al;
#pragma unroll
    for (int j = 0; j < 16; ++j) {
      bf16_t h = (bf16_t)fa[j];
      ah[j] = h;
      al[j] = (bf16_t)(fa[j] - (float)h);
    }
    // batch-load all B fragments for this K-block (immediate offsets)
    v16bf bh[NT], bl[NT];
#pragma unroll
    for (int t = 0; t < NT; ++t)
      bh[t] = *(const v16bf*)(Bbase + kb + t * 16 * K);
#pragma unroll
    for (int t = 0; t < NT; ++t)
      bl[t] = *(const v16bf*)(Bbase + kb + t * 16 * K + K * C);
    // round-robin WMMA issue: same-acc dependency distance = NT
#pragma unroll
    for (int t = 0; t < NT; ++t)
      acc[t] = __builtin_amdgcn_wmma_f32_16x16x32_bf16(false, ah, false, bh[t],
                                                       (short)0, acc[t], false, false);
#pragma unroll
    for (int t = 0; t < NT; ++t)
      acc[t] = __builtin_amdgcn_wmma_f32_16x16x32_bf16(false, al, false, bh[t],
                                                       (short)0, acc[t], false, false);
#pragma unroll
    for (int t = 0; t < NT; ++t)
      acc[t] = __builtin_amdgcn_wmma_f32_16x16x32_bf16(false, ah, false, bl[t],
                                                       (short)0, acc[t], false, false);
  }

  // C/D layout: VGPR r -> row = 8*half + r, col = lane&15
#pragma unroll
  for (int t = 0; t < NT; ++t) {
    int ocol = colbase + t * 16 + m;
    float b = BIAS ? bias[ocol] : 0.f;
#pragma unroll
    for (int r = 0; r < 8; ++r) {
      int orow = tile * 16 + half * 8 + r;
      Cout[(size_t)orow * C + ocol] = acc[t][r] + b;
    }
  }
}

// --- edge aggregation: out[dst] += w * x[src] (L2-resident atomics) ---------
// thread = (edge, 16-col chunk); 16 lanes cover 256 cols of one edge.
__global__ void agg_scatter(const float* __restrict__ x,
                            const float* __restrict__ ew,
                            const int*   __restrict__ esrc,
                            const int*   __restrict__ edst,
                            float* __restrict__ out, int nwork) {
  int t = blockIdx.x * blockDim.x + threadIdx.x;
  if (t >= nwork) return;
  int e  = t >> 4;
  int c0 = (t & 15) << 4;
  int s = esrc[e];
  int d = edst[e];
  float w = ew[e];
  const v4f* xp = (const v4f*)(x + (size_t)s * HDIM + c0);
  float*     op = out + (size_t)d * HDIM + c0;
  v4f x0 = xp[0], x1 = xp[1], x2 = xp[2], x3 = xp[3];
#pragma unroll
  for (int i = 0; i < 4; ++i) atomicAdd(op + i,      w * x0[i]);
#pragma unroll
  for (int i = 0; i < 4; ++i) atomicAdd(op + 4 + i,  w * x1[i]);
#pragma unroll
  for (int i = 0; i < 4; ++i) atomicAdd(op + 8 + i,  w * x2[i]);
#pragma unroll
  for (int i = 0; i < 4; ++i) atomicAdd(op + 12 + i, w * x3[i]);
}

// --- BN column stats: sum / sumsq over rows (coalesced, per-column threads) -
__global__ void bn_stats(const float* __restrict__ x, float* __restrict__ sums,
                         int mrows, int rpb) {
  int c  = threadIdx.x;                 // 256 columns
  int r0 = blockIdx.x * rpb;
  int r1 = r0 + rpb; if (r1 > mrows) r1 = mrows;
  float s = 0.f, s2 = 0.f;
  for (int r = r0; r < r1; ++r) {
    float v = x[(size_t)r * HDIM + c];
    s += v;
    s2 = fmaf(v, v, s2);
  }
  atomicAdd(&sums[c], s);
  atomicAdd(&sums[HDIM + c], s2);
}

__global__ void bn_finalize(const float* __restrict__ sums,
                            const float* __restrict__ gamma,
                            const float* __restrict__ beta,
                            float* __restrict__ scale,
                            float* __restrict__ shift, float invN) {
  int c = threadIdx.x;
  float mean = sums[c] * invN;
  float var  = sums[HDIM + c] * invN - mean * mean;
  float sc   = gamma[c] * rsqrtf(var + BN_EPS);
  scale[c] = sc;
  shift[c] = beta[c] - mean * sc;
}

extern "C" void kernel_launch(void* const* d_in, const int* in_sizes, int n_in,
                              void* d_out, int out_size, void* d_ws, size_t ws_size,
                              hipStream_t stream) {
  const float* features = (const float*)d_in[0];
  const float* edge_w   = (const float*)d_in[1];
  const float* W0       = (const float*)d_in[2];
  const float* b0       = (const float*)d_in[3];
  // d_in[4] agg_b0: cancels inside BN
  const float* gamma0   = (const float*)d_in[5];
  const float* beta0    = (const float*)d_in[6];
  const float* W1       = (const float*)d_in[7];
  // d_in[8] agg_b1: cancels inside BN
  const float* gamma1   = (const float*)d_in[9];
  const float* beta1    = (const float*)d_in[10];
  const float* W_out    = (const float*)d_in[11];
  const float* b_out    = (const float*)d_in[12];
  const int*   edge_src = (const int*)d_in[13];
  const int*   edge_dst = (const int*)d_in[14];

  // workspace layout
  char*   ws   = (char*)d_ws;
  float*  hpre = (float*)ws;                                 // [N,256] f32
  float*  aggb = hpre + (size_t)NNODES * HDIM;               // [N,256] f32
  bf16_t* wb   = (bf16_t*)(aggb + (size_t)NNODES * HDIM);    // bf16 weights
  bf16_t* w0h = wb;                  // lo at +128*256
  bf16_t* w1h = wb + 65536;          // lo at +256*256
  bf16_t* woh = wb + 196608;         // lo at +256*64
  float* stats   = (float*)(wb + 229376);                    // sum[256]+sumsq[256]
  float* bnscale = stats + 2 * HDIM;
  float* bnshift = bnscale + HDIM;

  const int nwork  = NEDGES * 16;
  const int aggGrd = (nwork + 255) / 256;
  const int rpb    = (NNODES + 255) / 256;

  // 1) weights -> transposed bf16 hi/lo
  prep_weights<<<448, 256, 0, stream>>>(W0, W1, W_out, wb);

  // 2) h0 = features @ W0 + b0
  gemm_wmma<DIN, HDIM, 8, false, true><<<dim3(NNODES / 16, 2), 32, 0, stream>>>(
      features, w0h, b0, nullptr, nullptr, hpre);

  // 3) agg0 = scatter-add over edges
  hipMemsetAsync(aggb, 0, (size_t)NNODES * HDIM * sizeof(float), stream);
  agg_scatter<<<aggGrd, 256, 0, stream>>>(hpre, edge_w, edge_src, edge_dst, aggb, nwork);

  // 4) BN0 stats
  hipMemsetAsync(stats, 0, 2 * HDIM * sizeof(float), stream);
  bn_stats<<<256, 256, 0, stream>>>(aggb, stats, NNODES, rpb);
  bn_finalize<<<1, 256, 0, stream>>>(stats, gamma0, beta0, bnscale, bnshift, 1.f / NNODES);

  // 5) h1 = relu(bn0(agg0)) @ W1   (BN+ReLU fused into A load)
  gemm_wmma<HDIM, HDIM, 8, true, false><<<dim3(NNODES / 16, 2), 32, 0, stream>>>(
      aggb, w1h, nullptr, bnscale, bnshift, hpre);

  // 6) agg1
  hipMemsetAsync(aggb, 0, (size_t)NNODES * HDIM * sizeof(float), stream);
  agg_scatter<<<aggGrd, 256, 0, stream>>>(hpre, edge_w, edge_src, edge_dst, aggb, nwork);

  // 7) BN1 stats
  hipMemsetAsync(stats, 0, 2 * HDIM * sizeof(float), stream);
  bn_stats<<<256, 256, 0, stream>>>(aggb, stats, NNODES, rpb);
  bn_finalize<<<1, 256, 0, stream>>>(stats, gamma1, beta1, bnscale, bnshift, 1.f / NNODES);

  // 8) out = relu(bn1(agg1)) @ W_out + b_out
  gemm_wmma<HDIM, EMBD, 4, true, true><<<dim3(NNODES / 16, 1), 32, 0, stream>>>(
      aggb, woh, b_out, bnscale, bnshift, (float*)d_out);
}
